// SwinTransformerBlock_17325898072404
// MI455X (gfx1250) — matrix operand order repaired and checked by reading, not compiled, not run on hardware
//
#include <hip/hip_runtime.h>
#include <hip/hip_bf16.h>

// ---------------------------------------------------------------------------
// Swin Transformer block for MI455X (gfx1250, wave32, WMMA bf16 16x16x32)
// B=32 H=W=64 C=512 NH=16 HD=32 WS=8 SHIFT=4 NW=64 N=64 HID=2048
// GEMMs stage A-slabs in LDS via global_load_async_to_lds_b128 (ASYNCcnt).
// ---------------------------------------------------------------------------

typedef __bf16 bf16_t;
typedef __attribute__((ext_vector_type(16))) bf16_t bf16x16;
typedef __attribute__((ext_vector_type(8)))  float  floatx8;

#define CDIM   512
#define NHEAD  16
#define HDIM   32
#define NTOK   64      // tokens per window
#define NWIN   64      // windows per image
#define HIDDEN 2048
#define MTOK   131072  // B*H*W

// float -> bf16, round-to-nearest-even (bit-exact, no libcall dependence)
__device__ __forceinline__ bf16_t f2bf(float f) {
  unsigned u = __float_as_uint(f);
  u += 0x7FFFu + ((u >> 16) & 1u);
  unsigned short hb = (unsigned short)(u >> 16);
  bf16_t out;
  __builtin_memcpy(&out, &hb, 2);
  return out;
}

union FragU { bf16x16 v; uint4 q[2]; };

// A fragment (16x32 bf16, row-major A, lda in elements).
// lane<16  : row = base+(lane&15), K chunks [k0..k0+7] and [k0+16..k0+23]
// lane>=16 : same rows,            K chunks [k0+8..15]  and [k0+24..31]
__device__ __forceinline__ bf16x16 load_a(const bf16_t* base, int ld, int row,
                                          int k0, int hl) {
  FragU f;
  const bf16_t* p = base + (size_t)row * ld + k0 + (hl ? 8 : 0);
  f.q[0] = *(const uint4*)(p);
  f.q[1] = *(const uint4*)(p + 16);
  return f.v;
}

// B fragment from B^T (N x K row-major, ld in elements).
// lane<16: K [k0..k0+7],[k0+16..k0+23]; lane>=16: K [k0+8..15],[k0+24..31]; row = n tile + (lane&15)
__device__ __forceinline__ bf16x16 load_bt(const bf16_t* base, int ld, int nrow,
                                           int k0, int hl) {
  FragU f;
  const bf16_t* p = base + (size_t)nrow * ld + k0 + hl * 8;
  f.q[0] = *(const uint4*)(p);
  f.q[1] = *(const uint4*)(p + 16);
  return f.v;
}

__device__ __forceinline__ floatx8 wmma_bf16(bf16x16 a, bf16x16 b, floatx8 c) {
  return __builtin_amdgcn_wmma_f32_16x16x32_bf16(false, a, false, b,
                                                 (short)0, c, false, false);
}

__device__ __forceinline__ int sgroup(int x) {  // shifted-window mask group
  return x < 56 ? 0 : (x < 60 ? 1 : 2);
}

// Async global->LDS copy of one 16B chunk (no VGPR data path, ASYNCcnt).
// lds_off is the byte offset inside this kernel's (single) __shared__ block,
// which the compiler places at LDS offset 0.
__device__ __forceinline__ void async_b128(unsigned lds_off, const bf16_t* g) {
  asm volatile("global_load_async_to_lds_b128 %0, %1, off"
               :: "v"(lds_off), "v"((unsigned long long)g)
               : "memory");
}

#define WAIT_ASYNC(n) asm volatile("s_wait_asynccnt %0" ::"i"(n) : "memory")

// ---------------------------------------------------------------------------
// Weight repack: w (K x N) f32 -> bt (N x K) bf16
// ---------------------------------------------------------------------------
__global__ __launch_bounds__(256) void repack_bt_kernel(
    const float* __restrict__ w, bf16_t* __restrict__ bt, int K, int N) {
  size_t i = (size_t)blockIdx.x * 256 + threadIdx.x;
  if (i >= (size_t)K * N) return;
  int n = (int)(i / K);
  int k = (int)(i % K);
  bt[i] = f2bf(w[(size_t)k * N + n]);
}

// ---------------------------------------------------------------------------
// Fused rel-pos bias + shifted-window mask table:
// addt[((wid*16 + h)*64 + key)*64 + query], query contiguous so the attention
// kernel loads each accumulator's 8 elements as two b128 vector loads.
// ---------------------------------------------------------------------------
__global__ __launch_bounds__(256) void bias_mask_kernel(
    const float* __restrict__ relb, float* __restrict__ addt) {
  const size_t i = (size_t)blockIdx.x * 256 + threadIdx.x;  // 4,194,304 total
  const int mq = (int)(i & 63);          // query token
  const int nk = (int)((i >> 6) & 63);   // key token
  const int h = (int)((i >> 12) & 15);
  const int wid = (int)(i >> 16);
  const int imq = mq >> 3, jmq = mq & 7;
  const int ink = nk >> 3, jnk = nk & 7;
  const int bidx = (imq - ink + 7) * 15 + (jmq - jnk + 7);
  const int wh = wid >> 3, ww = wid & 7;
  const int grow = sgroup(wh * 8 + imq) * 3 + sgroup(ww * 8 + jmq);
  const int gcol = sgroup(wh * 8 + ink) * 3 + sgroup(ww * 8 + jnk);
  float t = relb[bidx * NHEAD + h];
  if (grow != gcol) t -= 100.0f;
  addt[i] = t;
}

// ---------------------------------------------------------------------------
// LayerNorm (wave per token).  REMAP=true: fuse cyclic shift (-4,-4) +
// window partition into the bf16 output layout (bw, n, C).
// ---------------------------------------------------------------------------
template <bool REMAP>
__global__ __launch_bounds__(256) void ln_kernel(
    const float* __restrict__ x, const float* __restrict__ gamma,
    const float* __restrict__ beta, bf16_t* __restrict__ out) {
  const int lane = threadIdx.x & 31;
  const int wave = threadIdx.x >> 5;
  const size_t tok = (size_t)blockIdx.x * 8 + wave;
  const float* xp = x + tok * CDIM;
  float4 vals[4];
  float s = 0.f;
#pragma unroll
  for (int i = 0; i < 4; ++i) {
    vals[i] = ((const float4*)xp)[lane + i * 32];
    s += vals[i].x + vals[i].y + vals[i].z + vals[i].w;
  }
#pragma unroll
  for (int o = 16; o > 0; o >>= 1) s += __shfl_xor(s, o, 32);
  const float mu = s * (1.0f / CDIM);
  float vs = 0.f;
#pragma unroll
  for (int i = 0; i < 4; ++i) {
    float dx = vals[i].x - mu, dy = vals[i].y - mu;
    float dz = vals[i].z - mu, dw = vals[i].w - mu;
    vs += dx * dx + dy * dy + dz * dz + dw * dw;
  }
#pragma unroll
  for (int o = 16; o > 0; o >>= 1) vs += __shfl_xor(vs, o, 32);
  const float rstd = rsqrtf(vs * (1.0f / CDIM) + 1e-5f);

  bf16_t* op;
  if (REMAP) {
    const int bidx = (int)(tok >> 12);
    const int hw = (int)(tok & 4095);
    const int rs = ((hw >> 6) + 60) & 63;  // shifted row
    const int cs = ((hw & 63) + 60) & 63;  // shifted col
    const size_t bw = (size_t)bidx * NWIN + (rs >> 3) * 8 + (cs >> 3);
    const int n = (rs & 7) * 8 + (cs & 7);
    op = out + (bw * NTOK + n) * CDIM;
  } else {
    op = out + tok * CDIM;
  }
#pragma unroll
  for (int i = 0; i < 4; ++i) {
    const int c0 = (lane + i * 32) * 4;
    const float4 vv = vals[i];
    op[c0 + 0] = f2bf((vv.x - mu) * rstd * gamma[c0 + 0] + beta[c0 + 0]);
    op[c0 + 1] = f2bf((vv.y - mu) * rstd * gamma[c0 + 1] + beta[c0 + 1]);
    op[c0 + 2] = f2bf((vv.z - mu) * rstd * gamma[c0 + 2] + beta[c0 + 2]);
    op[c0 + 3] = f2bf((vv.w - mu) * rstd * gamma[c0 + 3] + beta[c0 + 3]);
  }
}

// ---------------------------------------------------------------------------
// Templated WMMA GEMM: C(MxN) = A(MxK,bf16) * BT(NxK,bf16)^T + bias, fused
// epilogue. Wave = 64x32 tile (8 accums), block = 8 waves = 128x128 tile.
// A slab (128x32 bf16 = 8KB) is double-buffered in LDS via async copies.
// MODE 0: QKV split  (q scaled, k row-major, v transposed)
// MODE 1: proj + window-reverse + unshift + residual -> f32 out
// MODE 2: fc1 + exact-erf GELU -> bf16 out
// MODE 3: fc2 + residual -> f32 out
// ---------------------------------------------------------------------------
template <int MODE>
__global__ __launch_bounds__(256) void gemm_wmma_kernel(
    const bf16_t* __restrict__ A, const bf16_t* __restrict__ BT,
    int M, int N, int K, const float* __restrict__ bias,
    float* outF, const float* xres,
    bf16_t* outQ, bf16_t* outK, bf16_t* outV, bf16_t* outB) {
  __shared__ bf16_t lds_buf[2 * 128 * 32];  // 2 x 8KB A slabs
  const int lane = threadIdx.x & 31;
  const int wave = threadIdx.x >> 5;
  const int r  = lane & 15;
  const int hl = lane >> 4;
  const int m0 = blockIdx.y * 128 + (wave >> 2) * 64;
  const int n0 = blockIdx.x * 128 + (wave & 3) * 32;

  // cooperative async A-slab copy: 256 threads x 2 b128 = 8KB
  const int srow = threadIdx.x >> 1;   // 0..127
  const int ssel = threadIdx.x & 1;    // which 32B half of the 64B row
  const int mblk = blockIdx.y * 128;
  const bf16_t* g0 = A + (size_t)(mblk + srow) * K + ssel * 16;
  const unsigned l0 = (unsigned)(srow * 64 + ssel * 32);

  // prologue: slab for k0=0 into buffer 0
  async_b128(l0, g0);
  async_b128(l0 + 16u, g0 + 8);

  floatx8 acc[4][2] = {};
  for (int k0 = 0; k0 < K; k0 += 32) {
    const int cur = (k0 >> 5) & 1;
    if (k0 + 32 < K) {  // issue next slab, wait only for the previous one
      const bf16_t* gn = g0 + k0 + 32;
      const unsigned ln = l0 + (unsigned)((cur ^ 1) * 8192);
      async_b128(ln, gn);
      async_b128(ln + 16u, gn + 8);
      WAIT_ASYNC(2);
    } else {
      WAIT_ASYNC(0);
    }
    __syncthreads();

    const bf16_t* lA = lds_buf + cur * 4096;
    bf16x16 af[4], bfr[2];
#pragma unroll
    for (int i = 0; i < 4; ++i)
      af[i] = load_a(lA, 32, (wave >> 2) * 64 + i * 16 + r, 0, hl);
#pragma unroll
    for (int j = 0; j < 2; ++j) bfr[j] = load_bt(BT, K, n0 + j * 16 + r, k0, hl);
    __builtin_prefetch(BT + (size_t)(n0 + r) * K + k0 + 32, 0, 1);
#pragma unroll
    for (int i = 0; i < 4; ++i)
#pragma unroll
      for (int j = 0; j < 2; ++j)
        acc[i][j] = wmma_bf16(af[i], bfr[j], acc[i][j]);
    __syncthreads();  // protect buffer cur before it is refilled next+1 iter
  }

#pragma unroll
  for (int i = 0; i < 4; ++i) {
#pragma unroll
    for (int j = 0; j < 2; ++j) {
      const int col = n0 + j * 16 + r;
      const int mb = m0 + i * 16 + hl * 8;
      const float bv = bias[col];
#pragma unroll
      for (int e = 0; e < 8; ++e) {
        const int m = mb + e;
        float v = acc[i][j][e] + bv;
        if (MODE == 0) {
          const int sec = col >> 9;  // 0:q 1:k 2:v
          const int cc = col & 511;
          const int hh = cc >> 5, d = cc & 31;
          const size_t gid = (size_t)(m >> 6) * NHEAD + hh;  // (bw, head)
          const int tokn = m & 63;
          if (sec == 0)
            outQ[(gid * NTOK + tokn) * HDIM + d] = f2bf(v * 0.17677669529663687f);
          else if (sec == 1)
            outK[(gid * NTOK + tokn) * HDIM + d] = f2bf(v);
          else
            outV[(gid * HDIM + d) * NTOK + tokn] = f2bf(v);  // v transposed
        } else if (MODE == 1) {
          const int bw = m >> 6, tokn = m & 63;
          const int b = bw >> 6, wid = bw & 63;
          const int rs = (wid >> 3) * 8 + (tokn >> 3);
          const int cs = (wid & 7) * 8 + (tokn & 7);
          const int rr = (rs + 4) & 63, c2 = (cs + 4) & 63;  // unshift
          const size_t idx = (((size_t)b * 4096) + rr * 64 + c2) * CDIM + col;
          outF[idx] = xres[idx] + v;
        } else if (MODE == 2) {
          const float gl = 0.5f * v * (1.0f + erff(v * 0.70710678118654752f));
          outB[(size_t)m * (size_t)N + col] = f2bf(gl);
        } else {
          const size_t idx = (size_t)m * (size_t)N + col;
          outF[idx] = xres[idx] + v;
        }
      }
    }
  }
}

// ---------------------------------------------------------------------------
// Windowed attention: one wave per (window, head), register-pressure-shaped:
// the 64x64 score matrix is processed in two 32-row halves (64 accumulator
// VGPRs live instead of 128), softmax+P-store releases each half before the
// next; P@v likewise produces 32 rows at a time. Keeps the kernel under the
// 256-VGPR window (no s_set_vgpr_msb, fewer WMMA->VALU hazard NOPs).
// ---------------------------------------------------------------------------
__global__ __launch_bounds__(128) void attn_kernel(
    const bf16_t* __restrict__ qb, const bf16_t* __restrict__ kb,
    const bf16_t* __restrict__ vtb, const float* __restrict__ addt,
    bf16_t* __restrict__ ob) {
  __shared__ bf16_t plds[4 * NTOK * NTOK];  // 32 KB: 64x64 bf16 per wave
  const int lane = threadIdx.x & 31;
  const int wave = threadIdx.x >> 5;
  const int r  = lane & 15;
  const int hl = lane >> 4;
  const size_t gid = (size_t)blockIdx.x * 4 + wave;  // bw*16 + head
  const size_t bw = gid >> 4;
  const int h = (int)(gid & 15);
  const bf16_t* q  = qb  + gid * (NTOK * HDIM);
  const bf16_t* k  = kb  + gid * (NTOK * HDIM);
  const bf16_t* vt = vtb + gid * (HDIM * NTOK);
  bf16_t* P = plds + wave * (NTOK * NTOK);
  const float* ab = addt + ((size_t)(bw & 63) * NHEAD + h) * (NTOK * NTOK);

  // k fragments stay resident across both row halves
  bf16x16 bk[4];
#pragma unroll
  for (int j = 0; j < 4; ++j) bk[j] = load_bt(k, HDIM, j * 16 + r, 0, hl);

#pragma unroll
  for (int ih = 0; ih < 2; ++ih) {  // rows [ih*32, ih*32+32)
    floatx8 s2[2][4];
#pragma unroll
    for (int i2 = 0; i2 < 2; ++i2) {
      const bf16x16 aq = load_a(q, HDIM, (ih * 2 + i2) * 16 + r, 0, hl);
      const floatx8 z = {};
#pragma unroll
      for (int j = 0; j < 4; ++j) s2[i2][j] = wmma_bf16(aq, bk[j], z);
    }
#pragma unroll
    for (int i2 = 0; i2 < 2; ++i2) {
      const int itile = ih * 2 + i2;
      // fused rel-pos bias + shift mask, vectorized on the C-fragment layout
#pragma unroll
      for (int j = 0; j < 4; ++j) {
        const float4* p =
            (const float4*)(ab + (j * 16 + r) * NTOK + itile * 16 + hl * 8);
        const float4 b0 = p[0], b1 = p[1];
        s2[i2][j][0] += b0.x; s2[i2][j][1] += b0.y;
        s2[i2][j][2] += b0.z; s2[i2][j][3] += b0.w;
        s2[i2][j][4] += b1.x; s2[i2][j][5] += b1.y;
        s2[i2][j][6] += b1.z; s2[i2][j][7] += b1.w;
      }
      // row softmax (row spans 16 lanes of this half x 4 col-tiles)
#pragma unroll
      for (int e = 0; e < 8; ++e) {
        float mx = fmaxf(fmaxf(s2[i2][0][e], s2[i2][1][e]),
                         fmaxf(s2[i2][2][e], s2[i2][3][e]));
#pragma unroll
        for (int o = 1; o < 16; o <<= 1) mx = fmaxf(mx, __shfl_xor(mx, o, 32));
        float sum = 0.f;
#pragma unroll
        for (int j = 0; j < 4; ++j) {
          float ev = __expf(s2[i2][j][e] - mx);
          s2[i2][j][e] = ev;
          sum += ev;
        }
#pragma unroll
        for (int o = 1; o < 16; o <<= 1) sum += __shfl_xor(sum, o, 32);
        const float inv = 1.0f / sum;
#pragma unroll
        for (int j = 0; j < 4; ++j) s2[i2][j][e] *= inv;
      }
      // restage P rows through LDS (C-fragment -> A-fragment reshape)
#pragma unroll
      for (int j = 0; j < 4; ++j)
#pragma unroll
        for (int e = 0; e < 8; ++e)
          P[(itile * 16 + hl * 8 + e) * NTOK + j * 16 + r] =
              f2bf(s2[i2][j][e]);
    }
  }
  __syncthreads();

  // O = P @ V, 32 output rows at a time (v arrives pre-transposed)
#pragma unroll
  for (int ih = 0; ih < 2; ++ih) {
    floatx8 o2[2][2] = {};
#pragma unroll
    for (int ks = 0; ks < 2; ++ks) {
      const bf16x16 vb0 = load_bt(vt, NTOK, r, ks * 32, hl);
      const bf16x16 vb1 = load_bt(vt, NTOK, 16 + r, ks * 32, hl);
#pragma unroll
      for (int i2 = 0; i2 < 2; ++i2) {
        const bf16x16 pa = load_a(P, NTOK, (ih * 2 + i2) * 16 + r, ks * 32, hl);
        o2[i2][0] = wmma_bf16(pa, vb0, o2[i2][0]);
        o2[i2][1] = wmma_bf16(pa, vb1, o2[i2][1]);
      }
    }
#pragma unroll
    for (int i2 = 0; i2 < 2; ++i2)
#pragma unroll
      for (int j = 0; j < 2; ++j)
#pragma unroll
        for (int e = 0; e < 8; ++e) {
          const int mrow = (ih * 2 + i2) * 16 + hl * 8 + e;
          const int d = j * 16 + r;
          ob[((size_t)bw * NTOK + mrow) * CDIM + h * HDIM + d] =
              f2bf(o2[i2][j][e]);
        }
  }
}

// ---------------------------------------------------------------------------
// Launcher
// ---------------------------------------------------------------------------
extern "C" void kernel_launch(void* const* d_in, const int* in_sizes, int n_in,
                              void* d_out, int out_size, void* d_ws,
                              size_t ws_size, hipStream_t stream) {
  (void)in_sizes; (void)n_in; (void)out_size; (void)ws_size;
  const float* x      = (const float*)d_in[0];
  const float* qkv_w  = (const float*)d_in[1];
  const float* qkv_b  = (const float*)d_in[2];
  const float* proj_w = (const float*)d_in[3];
  const float* proj_b = (const float*)d_in[4];
  const float* relb   = (const float*)d_in[5];
  const float* ln1_g  = (const float*)d_in[6];
  const float* ln1_b  = (const float*)d_in[7];
  const float* ln2_g  = (const float*)d_in[8];
  const float* ln2_b  = (const float*)d_in[9];
  const float* fc1_w  = (const float*)d_in[10];
  const float* fc1_b  = (const float*)d_in[11];
  const float* fc2_w  = (const float*)d_in[12];
  const float* fc2_b  = (const float*)d_in[13];
  float* out = (float*)d_out;

  char* ws = (char*)d_ws;
  size_t off = 0;
  auto take = [&](size_t bytes) -> char* {
    char* p = ws + off;
    off += (bytes + 255) & ~(size_t)255;
    return p;
  };
  const size_t M = MTOK;
  bf16_t* XW  = (bf16_t*)take(M * CDIM * 2);  // LN1 + shift + windows
  bf16_t* BT1 = (bf16_t*)take((size_t)1536 * 512 * 2);
  bf16_t* BT2 = (bf16_t*)take((size_t)512 * 512 * 2);
  bf16_t* BT3 = (bf16_t*)take((size_t)2048 * 512 * 2);
  bf16_t* BT4 = (bf16_t*)take((size_t)512 * 2048 * 2);
  bf16_t* Qb  = (bf16_t*)take(M * CDIM * 2);
  bf16_t* Kb  = (bf16_t*)take(M * CDIM * 2);
  bf16_t* VTb = (bf16_t*)take(M * CDIM * 2);
  bf16_t* Ob  = (bf16_t*)take(M * CDIM * 2);
  bf16_t* A2  = (bf16_t*)take(M * CDIM * 2);
  bf16_t* M1  = (bf16_t*)take(M * HIDDEN * 2);
  float*  ADDT = (float*)take((size_t)NWIN * NHEAD * NTOK * NTOK * 4);  // 16MB

  // weight repacks (f32 KxN -> bf16 NxK)
  repack_bt_kernel<<<(1536 * 512 + 255) / 256, 256, 0, stream>>>(qkv_w, BT1, 512, 1536);
  repack_bt_kernel<<<(512 * 512 + 255) / 256, 256, 0, stream>>>(proj_w, BT2, 512, 512);
  repack_bt_kernel<<<(2048 * 512 + 255) / 256, 256, 0, stream>>>(fc1_w, BT3, 512, 2048);
  repack_bt_kernel<<<(512 * 2048 + 255) / 256, 256, 0, stream>>>(fc2_w, BT4, 2048, 512);

  // fused bias+mask table (64 win x 16 heads x 64 x 64)
  bias_mask_kernel<<<(NWIN * NHEAD * NTOK * NTOK) / 256, 256, 0, stream>>>(
      relb, ADDT);

  // LN1 fused with cyclic shift + window partition
  ln_kernel<true><<<M / 8, 256, 0, stream>>>(x, ln1_g, ln1_b, XW);

  // QKV GEMM (131072 x 1536, K=512)
  gemm_wmma_kernel<0><<<dim3(1536 / 128, (int)(M / 128)), 256, 0, stream>>>(
      XW, BT1, (int)M, 1536, 512, qkv_b, nullptr, nullptr, Qb, Kb, VTb, nullptr);

  // windowed MHSA: 2048 windows x 16 heads, 4 (win,head) per block
  attn_kernel<<<(2048 * 16) / 4, 128, 0, stream>>>(Qb, Kb, VTb, ADDT, Ob);

  // proj GEMM + window reverse + unshift + residual -> x1 (in d_out)
  gemm_wmma_kernel<1><<<dim3(512 / 128, (int)(M / 128)), 256, 0, stream>>>(
      Ob, BT2, (int)M, 512, 512, proj_b, out, x, nullptr, nullptr, nullptr,
      nullptr);

  // LN2 on x1
  ln_kernel<false><<<M / 8, 256, 0, stream>>>(out, ln2_g, ln2_b, A2);

  // fc1 + exact GELU (131072 x 2048, K=512)
  gemm_wmma_kernel<2><<<dim3(2048 / 128, (int)(M / 128)), 256, 0, stream>>>(
      A2, BT3, (int)M, 2048, 512, fc1_b, nullptr, nullptr, nullptr, nullptr,
      nullptr, M1);

  // fc2 + residual (131072 x 512, K=2048) -> final output
  gemm_wmma_kernel<3><<<dim3(512 / 128, (int)(M / 128)), 256, 0, stream>>>(
      M1, BT4, (int)M, 512, 2048, fc2_b, out, out, nullptr, nullptr, nullptr,
      nullptr);
}
